// QuantumNAT_65481071396307
// MI455X (gfx1250) — matrix operand order repaired and checked by reading, not compiled, not run on hardware
//
#include <hip/hip_runtime.h>
#include <hip/hip_bf16.h>

typedef __attribute__((ext_vector_type(16))) _Float16 v16h;
typedef __attribute__((ext_vector_type(8)))  _Float16 v8h;
typedef __attribute__((ext_vector_type(8)))  float    v8f;

#define B_SAMPLES 8192
#define KDIM 784
#define KPAD 800   // 25 * 32
#define KSTEPS 25
#define BPREP_ELEMS (KSTEPS * 4 * 32 * 16)   // 51200 f16 = 102400 bytes

// ---------------------------------------------------------------------------
// Kernel 0: convert fc1_w [64,784] f32 -> f16 in WMMA B-fragment order:
//   bprep[ks][nt][lane][e], n = nt*16 + (lane&15),
//   k = ks*32 + (lane>=16 ? 16 : 0) + e, zero for k >= 784.
// Done once; every GEMM workgroup then streams it with async LDS copies.
// ---------------------------------------------------------------------------
__global__ __launch_bounds__(256) void wprep_kernel(
    const float* __restrict__ fc1w, _Float16* __restrict__ bprep)
{
    const int idx = blockIdx.x * 256 + threadIdx.x;
    if (idx >= BPREP_ELEMS) return;
    const int e    = idx & 15;
    const int lane = (idx >> 4) & 31;
    const int nt   = (idx >> 9) & 3;
    const int ks   = idx >> 11;
    const int n = nt * 16 + (lane & 15);
    const int k = ks * 32 + ((lane >> 4) ? 16 : 0) + e;
    const float v = (k < KDIM) ? fc1w[n * KDIM + k] : 0.0f;
    bprep[idx] = (_Float16)v;
}

// ---------------------------------------------------------------------------
// Kernel 1: conv1+relu+pool -> conv2+relu+pool, fully in LDS, one WG/sample.
// Writes flattened activations as f16 [B, KPAD] (K zero-padded 784->800).
// ---------------------------------------------------------------------------
__global__ __launch_bounds__(128) void front_conv_kernel(
    const float* __restrict__ x,
    const float* __restrict__ w1, const float* __restrict__ b1,
    const float* __restrict__ w2, const float* __restrict__ b2,
    _Float16* __restrict__ aflat)
{
    __shared__ float xs[784];
    __shared__ float w1s[72];
    __shared__ float b1s[8];
    __shared__ float w2s[1152];
    __shared__ float b2s[16];
    __shared__ float p1[8 * 14 * 14];

    const int tid = threadIdx.x;
    const int b   = blockIdx.x;
    const float* xb = x + (size_t)b * 784;

    for (int i = tid; i < 784;  i += 128) xs[i]  = xb[i];
    for (int i = tid; i < 72;   i += 128) w1s[i] = w1[i];
    for (int i = tid; i < 1152; i += 128) w2s[i] = w2[i];
    if (tid < 8)  b1s[tid] = b1[tid];
    if (tid < 16) b2s[tid] = b2[tid];
    __syncthreads();

    // conv1 (SAME, 3x3, 1->8) + relu + maxpool2 -> p1[8][14][14]
    for (int o = tid; o < 8 * 14 * 14; o += 128) {
        const int ch = o / 196, rem = o % 196, pi = rem / 14, pj = rem % 14;
        const float* wc = &w1s[ch * 9];
        float m = -1e30f;
        #pragma unroll
        for (int di = 0; di < 2; ++di)
        #pragma unroll
        for (int dj = 0; dj < 2; ++dj) {
            const int ci = 2 * pi + di, cj = 2 * pj + dj;
            float acc = b1s[ch];
            #pragma unroll
            for (int ki = 0; ki < 3; ++ki) {
                const int yi = ci + ki - 1;
                if (yi < 0 || yi >= 28) continue;
                #pragma unroll
                for (int kj = 0; kj < 3; ++kj) {
                    const int yj = cj + kj - 1;
                    if (yj < 0 || yj >= 28) continue;
                    acc += xs[yi * 28 + yj] * wc[ki * 3 + kj];
                }
            }
            m = fmaxf(m, acc);
        }
        p1[o] = fmaxf(m, 0.0f);   // relu(max) == max(relu)
    }
    __syncthreads();

    // conv2 (SAME, 3x3, 8->16) + relu + maxpool2 -> flat[784], pad to 800
    for (int o = tid; o < KPAD; o += 128) {
        float v = 0.0f;
        if (o < KDIM) {
            const int ch = o / 49, rem = o % 49, pi = rem / 7, pj = rem % 7;
            float m = -1e30f;
            #pragma unroll
            for (int di = 0; di < 2; ++di)
            #pragma unroll
            for (int dj = 0; dj < 2; ++dj) {
                const int ci = 2 * pi + di, cj = 2 * pj + dj;
                float acc = b2s[ch];
                for (int ic = 0; ic < 8; ++ic) {
                    const float* wc = &w2s[(ch * 8 + ic) * 9];
                    const float* pc = &p1[ic * 196];
                    #pragma unroll
                    for (int ki = 0; ki < 3; ++ki) {
                        const int yi = ci + ki - 1;
                        if (yi < 0 || yi >= 14) continue;
                        #pragma unroll
                        for (int kj = 0; kj < 3; ++kj) {
                            const int yj = cj + kj - 1;
                            if (yj < 0 || yj >= 14) continue;
                            acc += pc[yi * 14 + yj] * wc[ki * 3 + kj];
                        }
                    }
                }
                m = fmaxf(m, acc);
            }
            v = fmaxf(m, 0.0f);
        }
        aflat[(size_t)b * KPAD + o] = (_Float16)v;
    }
}

// ---------------------------------------------------------------------------
// Kernel 2: 4-qubit state-vector simulation, one thread per sample.
// ---------------------------------------------------------------------------
__device__ __forceinline__ float2 cmul(float2 a, float2 b) {
    return make_float2(a.x * b.x - a.y * b.y, a.x * b.y + a.y * b.x);
}
__device__ __forceinline__ float2 cadd(float2 a, float2 b) {
    return make_float2(a.x + b.x, a.y + b.y);
}
__device__ __forceinline__ void apply1q(float2* s, int wire,
                                        float2 u00, float2 u01, float2 u10, float2 u11)
{
    const int bit = 1 << (3 - wire);   // psi[B,2,2,2,2]: wire w is bit (3-w)
    #pragma unroll
    for (int i = 0; i < 16; ++i) {
        if (i & bit) continue;
        const float2 a = s[i], c = s[i | bit];
        s[i]       = cadd(cmul(u00, a), cmul(u01, c));
        s[i | bit] = cadd(cmul(u10, a), cmul(u11, c));
    }
}
__device__ __forceinline__ void apply_ry(float2* s, int w, float t) {
    const float c = __cosf(0.5f * t), sn = __sinf(0.5f * t);
    apply1q(s, w, make_float2(c, 0), make_float2(-sn, 0), make_float2(sn, 0), make_float2(c, 0));
}
__device__ __forceinline__ void apply_rz(float2* s, int w, float t) {
    const float c = __cosf(0.5f * t), sn = __sinf(0.5f * t);
    apply1q(s, w, make_float2(c, -sn), make_float2(0, 0), make_float2(0, 0), make_float2(c, sn));
}
__device__ __forceinline__ void apply_rx(float2* s, int w, float t) {
    const float c = __cosf(0.5f * t), sn = __sinf(0.5f * t);
    apply1q(s, w, make_float2(c, 0), make_float2(0, -sn), make_float2(0, -sn), make_float2(c, 0));
}
__device__ __forceinline__ void apply_cx(float2* s, int cw, int tw) {
    const int bc = 1 << (3 - cw), bt = 1 << (3 - tw);
    #pragma unroll
    for (int i = 0; i < 16; ++i) {
        if ((i & bc) && !(i & bt)) {
            const float2 tmp = s[i]; s[i] = s[i | bt]; s[i | bt] = tmp;
        }
    }
}

__global__ __launch_bounds__(128) void quantum_kernel(
    const float* __restrict__ x,
    const float* __restrict__ cp,   // conv_params[12]
    const float* __restrict__ pp,   // pool_params[12]
    float* __restrict__ qout)       // [B,4]
{
    const int b = blockIdx.x * blockDim.x + threadIdx.x;
    if (b >= B_SAMPLES) return;
    const float* xb = x + (size_t)b * 784;

    // avgpool 6x6 stride 6 VALID -> pooled[16], k = ry*4 + rx
    float pooled[16];
    #pragma unroll
    for (int k = 0; k < 16; ++k) {
        const int r0 = (k >> 2) * 6, c0 = (k & 3) * 6;
        float ssum = 0.0f;
        for (int i = 0; i < 6; ++i)
            for (int j = 0; j < 6; ++j)
                ssum += xb[(r0 + i) * 28 + (c0 + j)];
        pooled[k] = ssum * (1.0f / 36.0f);
    }

    float2 st[16];
    #pragma unroll
    for (int i = 0; i < 16; ++i) st[i] = make_float2(0.0f, 0.0f);
    st[0] = make_float2(1.0f, 0.0f);

    // encoder: 4x4_ryzxy
    #pragma unroll
    for (int k = 0; k < 16; ++k) {
        const int w = k & 3;
        if      (k < 4)  apply_ry(st, w, pooled[k]);
        else if (k < 8)  apply_rz(st, w, pooled[k]);
        else if (k < 12) apply_rx(st, w, pooled[k]);
        else             apply_ry(st, w, pooled[k]);
    }
    // QLayer conv block
    apply_rz(st, 0, cp[0]); apply_ry(st, 1, cp[1]); apply_cx(st, 1, 0);
    apply_rz(st, 2, cp[3]); apply_ry(st, 3, cp[4]); apply_cx(st, 3, 2);
    // QLayer pool block
    apply_rz(st, 0, pp[0]); apply_ry(st, 1, pp[1]); apply_cx(st, 0, 1);
    apply_rz(st, 2, pp[3]); apply_ry(st, 3, pp[4]); apply_cx(st, 2, 3);

    // PauliZ expectations
    #pragma unroll
    for (int w = 0; w < 4; ++w) {
        float e = 0.0f;
        #pragma unroll
        for (int i = 0; i < 16; ++i) {
            const float p = st[i].x * st[i].x + st[i].y * st[i].y;
            e += ((i >> (3 - w)) & 1) ? -p : p;
        }
        qout[(size_t)b * 4 + w] = e;
    }
}

// ---------------------------------------------------------------------------
// Kernel 3: fc1 via v_wmma_f32_16x16x32_f16, fused bias+relu+fc2.
// Grid: 128 WGs of 128 threads (4 waves). Wave w owns a 16-row M-tile, all
// of N=64 (4 accumulators). Prologue: the pre-swizzled B matrix (100 KB) is
// streamed to LDS with global_load_async_to_lds_b128 (ASYNCcnt path), then
// the 25-step K loop runs barrier-free: 2 global b128 A loads, 8 ds b128 B
// loads, 4 WMMAs per step. The f32 epilogue tile reuses the same LDS.
// ---------------------------------------------------------------------------
__global__ __launch_bounds__(128) void fc_wmma_kernel(
    const _Float16* __restrict__ aflat,   // [B, KPAD] f16
    const _Float16* __restrict__ bprep,   // [25][4][32][16] f16 fragments
    const float* __restrict__ fc1b,       // [64]
    const float* __restrict__ fc2w,       // [4, 64]
    const float* __restrict__ fc2b,       // [4]
    float* __restrict__ fcout)            // [B, 4]
{
    // 102400 B for B-fragments; reused (>=16640 B) for the 64x65 f32 tile.
    __shared__ __align__(32) char smem[BPREP_ELEMS * 2];
    _Float16* bsmem = (_Float16*)smem;
    float*    ctile = (float*)smem;       // [64][65] after the K loop

    const int tid  = threadIdx.x;
    const int wave = tid >> 5;
    const int lane = tid & 31;
    const int hiHalf = lane >> 4;         // 0: lanes 0-15, 1: lanes 16-31
    const int m0 = blockIdx.x * 64 + wave * 16;

    // --- Stage pre-swizzled B into LDS with async global->LDS copies -------
    {
        const unsigned lds_base = (unsigned)(size_t)smem;  // low 32 bits = LDS offset
        #pragma unroll 1
        for (int i = tid; i < BPREP_ELEMS * 2 / 16; i += 128) {   // 6400 x 16B
            const unsigned off = (unsigned)i * 16u;
            asm volatile(
                "global_load_async_to_lds_b128 %0, %1, %2"
                :
                : "v"(lds_base + off), "v"(off), "s"(bprep)
                : "memory");
        }
        asm volatile("s_wait_asynccnt 0x0" ::: "memory");
    }
    __syncthreads();

    // --- A fragment base (16-bit A 16x32, contiguous k = 16*half + e) -------
    const int row = m0 + (lane & 15);
    const _Float16* abase = aflat + (size_t)row * KPAD + (hiHalf ? 16 : 0);

    v8f acc[4] = {};

    for (int ks = 0; ks < KSTEPS; ++ks) {
        const int kb = ks * 32;
        const v8h lo = *(const v8h*)(abase + kb);
        const v8h hi = *(const v8h*)(abase + kb + 8);
        const v16h a = __builtin_shufflevector(lo, hi,
            0, 1, 2, 3, 4, 5, 6, 7, 8, 9, 10, 11, 12, 13, 14, 15);
        if (ks + 1 < KSTEPS) __builtin_prefetch(abase + kb + 32, 0, 1);

        #pragma unroll
        for (int nt = 0; nt < 4; ++nt) {
            const v16h bf = *(const v16h*)(bsmem + (((ks * 4 + nt) * 32) + lane) * 16);
            acc[nt] = __builtin_amdgcn_wmma_f32_16x16x32_f16(
                /*neg_a=*/false, a, /*neg_b=*/false, bf,
                /*c_mod=*/(short)0, acc[nt], /*reuse_a=*/false, /*reuse_b=*/false);
        }
    }
    __syncthreads();   // all waves done reading bsmem; LDS reused as ctile

    // Epilogue: bias + relu, scatter C fragments (f32 C/D layout) to LDS tile.
    #pragma unroll
    for (int nt = 0; nt < 4; ++nt) {
        #pragma unroll
        for (int r = 0; r < 8; ++r) {
            const int mr = r + (hiHalf ? 8 : 0);
            const int n  = nt * 16 + (lane & 15);
            const float v = acc[nt][r] + fc1b[n];
            ctile[(wave * 16 + mr) * 65 + n] = fmaxf(v, 0.0f);
        }
    }
    __syncthreads();

    // fc2: 64 rows x 4 outputs, 2 outputs per thread.
    for (int o = tid; o < 256; o += 128) {
        const int r = o >> 2, j = o & 3;
        float dot = fc2b[j];
        #pragma unroll
        for (int n = 0; n < 64; ++n) dot += ctile[r * 65 + n] * fc2w[j * 64 + n];
        fcout[(size_t)(blockIdx.x * 64 + r) * 4 + j] = dot;
    }
}

// ---------------------------------------------------------------------------
// Kernel 4: batch statistics (sum, sum of squares) for fc[:,0..3], q[:,0..3].
// stats layout: [0..3]=sum_fc [4..7]=ss_fc [8..11]=sum_q [12..15]=ss_q
// ---------------------------------------------------------------------------
__global__ __launch_bounds__(256) void stats_kernel(
    const float* __restrict__ fc, const float* __restrict__ q,
    float* __restrict__ stats)
{
    __shared__ float sbuf[256];
    const int tid = threadIdx.x;
    float acc[16];
    #pragma unroll
    for (int i = 0; i < 16; ++i) acc[i] = 0.0f;

    for (int r = tid; r < B_SAMPLES; r += 256) {
        #pragma unroll
        for (int j = 0; j < 4; ++j) {
            const float vf = fc[r * 4 + j], vq = q[r * 4 + j];
            acc[j]      += vf;  acc[4 + j]  += vf * vf;
            acc[8 + j]  += vq;  acc[12 + j] += vq * vq;
        }
    }
    for (int qi = 0; qi < 16; ++qi) {
        sbuf[tid] = acc[qi];
        __syncthreads();
        for (int s = 128; s > 0; s >>= 1) {
            if (tid < s) sbuf[tid] += sbuf[tid + s];
            __syncthreads();
        }
        if (tid == 0) stats[qi] = sbuf[0];
        __syncthreads();
    }
}

// ---------------------------------------------------------------------------
// Kernel 5: apply BN (training-mode, biased var) to both branches and sum.
// ---------------------------------------------------------------------------
__global__ __launch_bounds__(256) void apply_bn_kernel(
    const float* __restrict__ fc, const float* __restrict__ q,
    const float* __restrict__ stats,
    const float* __restrict__ g, const float* __restrict__ be,
    float* __restrict__ out)
{
    const int idx = blockIdx.x * 256 + threadIdx.x;
    if (idx >= B_SAMPLES * 4) return;
    const int j = idx & 3;
    const float invB = 1.0f / (float)B_SAMPLES;
    const float mf = stats[j] * invB;
    const float vf = stats[4 + j] * invB - mf * mf;
    const float mq = stats[8 + j] * invB;
    const float vq = stats[12 + j] * invB - mq * mq;
    const float rf = rsqrtf(vf + 1e-5f);
    const float rq = rsqrtf(vq + 1e-5f);
    const float co = g[j] * (fc[idx] - mf) * rf + be[j];
    const float qo = g[j] * (q[idx]  - mq) * rq + be[j];
    out[idx] = co + qo;
}

// ---------------------------------------------------------------------------
extern "C" void kernel_launch(void* const* d_in, const int* in_sizes, int n_in,
                              void* d_out, int out_size, void* d_ws, size_t ws_size,
                              hipStream_t stream)
{
    const float* x   = (const float*)d_in[0];
    const float* c1w = (const float*)d_in[1];
    const float* c1b = (const float*)d_in[2];
    const float* c2w = (const float*)d_in[3];
    const float* c2b = (const float*)d_in[4];
    const float* f1w = (const float*)d_in[5];
    const float* f1b = (const float*)d_in[6];
    const float* f2w = (const float*)d_in[7];
    const float* f2b = (const float*)d_in[8];
    const float* g   = (const float*)d_in[9];
    const float* be  = (const float*)d_in[10];
    const float* cp  = (const float*)d_in[11];
    const float* pp  = (const float*)d_in[12];
    float* out = (float*)d_out;

    char* ws = (char*)d_ws;
    _Float16* aflat = (_Float16*)ws;                       // 8192*800*2 = 13,107,200 B
    float* qbuf  = (float*)(ws + 13107200);                // 131,072 B
    float* fcbuf = (float*)(ws + 13107200 + 131072);       // 131,072 B
    _Float16* bprep = (_Float16*)(ws + 13107200 + 2 * 131072);          // 102,400 B
    float* stats = (float*)(ws + 13107200 + 2 * 131072 + 102400);       // 64 B

    wprep_kernel<<<(BPREP_ELEMS + 255) / 256, 256, 0, stream>>>(f1w, bprep);
    front_conv_kernel<<<B_SAMPLES, 128, 0, stream>>>(x, c1w, c1b, c2w, c2b, aflat);
    quantum_kernel<<<B_SAMPLES / 128, 128, 0, stream>>>(x, cp, pp, qbuf);
    fc_wmma_kernel<<<B_SAMPLES / 64, 128, 0, stream>>>(aflat, bprep, f1b, f2w, f2b, fcbuf);
    stats_kernel<<<1, 256, 0, stream>>>(fcbuf, qbuf, stats);
    apply_bn_kernel<<<(B_SAMPLES * 4 + 255) / 256, 256, 0, stream>>>(fcbuf, qbuf, stats, g, be, out);
}
